// HardTripletLoss_27650999452613
// MI455X (gfx1250) — compile-verified, matrix operand order checked by reading, not run on hardware
//
#include <hip/hip_runtime.h>
#include <hip/hip_bf16.h>
#include <math.h>

typedef __attribute__((ext_vector_type(16))) _Float16 v16h;
typedef __attribute__((ext_vector_type(8)))  _Float16 v8h;
typedef __attribute__((ext_vector_type(8)))  float    v8f;

#define CDIM 256
#define HC 60
#define WC 80
#define BK 32        // K-slab per pipeline stage
#define LSTR 40      // padded LDS row stride (halfs): 80B rows -> conflict-free b128 reads

// ---------------------------------------------------------------------------
// CDNA5 async global->LDS copy of 16 bytes (tracked by ASYNCcnt, no VGPR data).
// ISA: GLOBAL_LOAD_ASYNC_TO_LDS_B128  vdst(LDS addr VGPR), vaddr(64b), off
// ---------------------------------------------------------------------------
__device__ __forceinline__ void htl_async_b128(const _Float16* g, unsigned lds_off) {
    asm volatile("global_load_async_to_lds_b128 %0, %1, off"
                 :: "v"(lds_off), "v"(g) : "memory");
}
#define HTL_WAIT_ASYNC(n) asm volatile("s_wait_asynccnt %0" :: "n"(n) : "memory")

__device__ __forceinline__ unsigned htl_lds_off(const _Float16* p) {
    return (unsigned)(unsigned long long)
        (__attribute__((address_space(3))) const _Float16*)p;  // low 32b = LDS byte addr
}

// Fragment from an LDS tile row (stride LSTR), matching the 16-bit 16x32
// WMMA layout: v16h[0..7] = K[hi*8 .. +7], v16h[8..15] = K[16+hi*8 .. +7].
__device__ __forceinline__ v16h htl_lds_frag(const _Float16* row, int hi) {
    v8h lo = *reinterpret_cast<const v8h*>(row + hi * 8);
    v8h hh = *reinterpret_cast<const v8h*>(row + hi * 8 + 16);
    v16h a;
    #pragma unroll
    for (int t = 0; t < 8; ++t) { a[t] = lo[t]; a[8 + t] = hh[t]; }
    return a;
}

// f32 -> f16 fragment load (fallback path only).
__device__ __forceinline__ v16h htl_load_frag_f32(const float* __restrict__ row, int k0) {
    const float4* p = reinterpret_cast<const float4*>(row + k0);
    const float4* q = reinterpret_cast<const float4*>(row + k0 + 16);
    float4 f0 = p[0], f1 = p[1];
    float4 g0 = q[0], g1 = q[1];
    v16h a;
    a[0]  = (_Float16)f0.x; a[1]  = (_Float16)f0.y; a[2]  = (_Float16)f0.z; a[3]  = (_Float16)f0.w;
    a[4]  = (_Float16)f1.x; a[5]  = (_Float16)f1.y; a[6]  = (_Float16)f1.z; a[7]  = (_Float16)f1.w;
    a[8]  = (_Float16)g0.x; a[9]  = (_Float16)g0.y; a[10] = (_Float16)g0.z; a[11] = (_Float16)g0.w;
    a[12] = (_Float16)g1.x; a[13] = (_Float16)g1.y; a[14] = (_Float16)g1.z; a[15] = (_Float16)g1.w;
    return a;
}

// ---------------------------------------------------------------------------
// One-shot f32 -> f16 staging: 8 elements per thread, b128 in / b128 out.
// ---------------------------------------------------------------------------
__global__ __launch_bounds__(256) void htl_f32_to_f16(const float* __restrict__ src,
                                                      _Float16* __restrict__ dst,
                                                      long n) {
    long i = ((long)blockIdx.x * 256 + threadIdx.x) * 8;
    if (i >= n) return;
    float4 f0 = *reinterpret_cast<const float4*>(src + i);
    float4 f1 = *reinterpret_cast<const float4*>(src + i + 4);
    v8h h;
    h[0] = (_Float16)f0.x; h[1] = (_Float16)f0.y; h[2] = (_Float16)f0.z; h[3] = (_Float16)f0.w;
    h[4] = (_Float16)f1.x; h[5] = (_Float16)f1.y; h[6] = (_Float16)f1.z; h[7] = (_Float16)f1.w;
    *reinterpret_cast<v8h*>(dst + i) = h;
}

__global__ __launch_bounds__(256) void htl_init(unsigned int* __restrict__ neg_bits, int n) {
    int i = blockIdx.x * 256 + threadIdx.x;
    if (i < n) neg_bits[i] = 0x7F800000u;  // +inf
}

// ---------------------------------------------------------------------------
// Row squared norms of both descriptor matrices. One wave32 per row.
// ---------------------------------------------------------------------------
__global__ __launch_bounds__(256) void htl_sqnorm(const float* __restrict__ a,
                                                  const float* __restrict__ b,
                                                  float* __restrict__ a_sq,
                                                  float* __restrict__ b_sq,
                                                  int N, int M) {
    int wave = threadIdx.x >> 5, lane = threadIdx.x & 31;
    int row = blockIdx.x * 8 + wave;
    if (row >= N + M) return;
    const float* src = (row < N) ? a : b;
    float* dst       = (row < N) ? a_sq : b_sq;
    int r            = (row < N) ? row : row - N;
    const float* p = src + (size_t)r * CDIM;
    float s = 0.f;
    #pragma unroll
    for (int c = 0; c < CDIM / 32; ++c) {
        float v = p[lane + 32 * c];
        s += v * v;
    }
    #pragma unroll
    for (int o = 16; o > 0; o >>= 1) s += __shfl_xor(s, o, 32);
    if (lane == 0) dst[r] = s;
}

// ---------------------------------------------------------------------------
// Shared epilogue: distance + neighbour mask + cross-lane row-min + atomicMin.
// D layout: lane (lr, hi), VGPR r -> row = i0 + 8*hi + r, col = j0 + 16*t + lr
// ---------------------------------------------------------------------------
__device__ __forceinline__ void htl_epilogue(
    v8f acc0, v8f acc1, v8f acc2, v8f acc3,
    const float* __restrict__ a_sq, const float* __restrict__ b_sq,
    const float* __restrict__ wkp1, const float* __restrict__ kp2,
    unsigned int* __restrict__ neg_bits, int i0, int j0, int hi, int lr) {
    union { v8f v; float f[8]; } U[4];
    U[0].v = acc0; U[1].v = acc1; U[2].v = acc2; U[3].v = acc3;

    float wy_[8], wx_[8], as_[8], mn[8];
    #pragma unroll
    for (int r = 0; r < 8; ++r) {
        int i = i0 + 8 * hi + r;
        wy_[r] = wkp1[2 * i];
        wx_[r] = wkp1[2 * i + 1];
        as_[r] = a_sq[i];
        mn[r]  = 3.4e38f;
    }
    const float thr  = 2.0f * 5.656854249492381f + 0.1f;  // 2*sqrt(GRID^2/2)+0.1
    const float thr2 = thr * thr;

    #pragma unroll
    for (int t = 0; t < 4; ++t) {
        int j = j0 + t * 16 + lr;
        float ky = kp2[2 * j], kx = kp2[2 * j + 1];
        float bs = b_sq[j];
        #pragma unroll
        for (int r = 0; r < 8; ++r) {
            float d = sqrtf(fmaxf(as_[r] + bs - 2.0f * U[t].f[r], 0.0f));
            float dy = wy_[r] - ky, dx = wx_[r] - kx;
            if (dy * dy + dx * dx <= thr2) d += 5.0f;
            mn[r] = fminf(mn[r], d);
        }
    }
    #pragma unroll
    for (int o = 1; o < 16; o <<= 1) {
        #pragma unroll
        for (int r = 0; r < 8; ++r)
            mn[r] = fminf(mn[r], __shfl_xor(mn[r], o, 32));
    }
    if (lr == 0) {
        #pragma unroll
        for (int r = 0; r < 8; ++r)
            atomicMin(neg_bits + (i0 + 8 * hi + r), __float_as_uint(mn[r]));
    }
}

// ---------------------------------------------------------------------------
// STAGED GEMM with async global->LDS double buffering, fully unrolled K
// pipeline (buf is compile-time constant per step -> accumulators stay
// pinned, LDS addresses fold into ds_load offsets, no v_mov shuffling).
// Block = 128 threads = 4 wave32s; block tile 64x64; wave tile 16x64.
// ---------------------------------------------------------------------------
__global__ __launch_bounds__(128) void htl_gemm_min_lds(
    const _Float16* __restrict__ ah, const _Float16* __restrict__ bh,
    const float* __restrict__ a_sq, const float* __restrict__ b_sq,
    const float* __restrict__ wkp1, const float* __restrict__ kp2,
    unsigned int* __restrict__ neg_bits) {
    __shared__ _Float16 As[2][64 * LSTR];
    __shared__ _Float16 Bs[2][64 * LSTR];

    int tid  = threadIdx.x;
    int wave = tid >> 5;
    int lane = tid & 31;
    int hi   = lane >> 4;
    int lr   = lane & 15;
    int i0 = blockIdx.y * 64;
    int j0 = blockIdx.x * 64;

    const _Float16* Ag = ah + (size_t)i0 * CDIM;
    const _Float16* Bg = bh + (size_t)j0 * CDIM;

    // Per-thread staging layout (computed once): thread covers rows
    // srow0 (=tid/4) and srow1 (=srow0+32), 16B segment `seg` of each row.
    int srow0 = tid >> 2;
    int srow1 = srow0 + 32;
    int seg   = tid & 3;
    const _Float16* gA0 = Ag + (size_t)srow0 * CDIM + seg * 8;
    const _Float16* gA1 = Ag + (size_t)srow1 * CDIM + seg * 8;
    const _Float16* gB0 = Bg + (size_t)srow0 * CDIM + seg * 8;
    const _Float16* gB1 = Bg + (size_t)srow1 * CDIM + seg * 8;
    unsigned la0[2] = { htl_lds_off(&As[0][srow0 * LSTR + seg * 8]),
                        htl_lds_off(&As[1][srow0 * LSTR + seg * 8]) };
    unsigned la1[2] = { htl_lds_off(&As[0][srow1 * LSTR + seg * 8]),
                        htl_lds_off(&As[1][srow1 * LSTR + seg * 8]) };
    unsigned lb0[2] = { htl_lds_off(&Bs[0][srow0 * LSTR + seg * 8]),
                        htl_lds_off(&Bs[1][srow0 * LSTR + seg * 8]) };
    unsigned lb1[2] = { htl_lds_off(&Bs[0][srow1 * LSTR + seg * 8]),
                        htl_lds_off(&Bs[1][srow1 * LSTR + seg * 8]) };

    auto stage = [&](int k, int b) {
        htl_async_b128(gA0 + k, la0[b]);
        htl_async_b128(gA1 + k, la1[b]);
        htl_async_b128(gB0 + k, lb0[b]);
        htl_async_b128(gB1 + k, lb1[b]);
    };

    // Prologue: stage slab 0 into buffer 0.
    stage(0, 0);

    v8f acc0 = {}, acc1 = {}, acc2 = {}, acc3 = {};
    const int NK = CDIM / BK;  // 8

    #pragma unroll
    for (int kk = 0; kk < NK; ++kk) {
        const int buf = kk & 1;
        if (kk + 1 < NK) {
            stage((kk + 1) * BK, buf ^ 1);  // +4 async ops
            HTL_WAIT_ASYNC(4);              // 4 oldest (= current buffer) done, in order
        } else {
            HTL_WAIT_ASYNC(0);
        }
        __syncthreads();                    // staged data visible to all waves

        const _Float16* ar = &As[buf][(wave * 16 + lr) * LSTR];
        v16h af = htl_lds_frag(ar, hi);
        v16h b0 = htl_lds_frag(&Bs[buf][( 0 + lr) * LSTR], hi);
        v16h b1 = htl_lds_frag(&Bs[buf][(16 + lr) * LSTR], hi);
        v16h b2 = htl_lds_frag(&Bs[buf][(32 + lr) * LSTR], hi);
        v16h b3 = htl_lds_frag(&Bs[buf][(48 + lr) * LSTR], hi);
        acc0 = __builtin_amdgcn_wmma_f32_16x16x32_f16(false, af, false, b0, (short)0, acc0, false, false);
        acc1 = __builtin_amdgcn_wmma_f32_16x16x32_f16(false, af, false, b1, (short)0, acc1, false, false);
        acc2 = __builtin_amdgcn_wmma_f32_16x16x32_f16(false, af, false, b2, (short)0, acc2, false, false);
        acc3 = __builtin_amdgcn_wmma_f32_16x16x32_f16(false, af, false, b3, (short)0, acc3, false, false);

        __syncthreads();                    // all waves done with buf before restage
    }

    htl_epilogue(acc0, acc1, acc2, acc3, a_sq, b_sq, wkp1, kp2, neg_bits,
                 i0 + wave * 16, j0, hi, lr);
}

// ---------------------------------------------------------------------------
// FALLBACK GEMM (only if ws_size can't hold the f16 staging buffers):
// converts f32 -> f16 in the loop, direct global loads.
// ---------------------------------------------------------------------------
__global__ __launch_bounds__(128) void htl_gemm_min_f(
    const float* __restrict__ adesc, const float* __restrict__ bdesc,
    const float* __restrict__ a_sq, const float* __restrict__ b_sq,
    const float* __restrict__ wkp1, const float* __restrict__ kp2,
    unsigned int* __restrict__ neg_bits) {
    int wave = threadIdx.x >> 5;
    int lane = threadIdx.x & 31;
    int hi   = lane >> 4;
    int lr   = lane & 15;
    int i0 = blockIdx.y * 64 + wave * 16;
    int j0 = blockIdx.x * 64;

    const float* arow  = adesc + (size_t)(i0 + lr) * CDIM;
    const float* brow0 = bdesc + (size_t)(j0 +  0 + lr) * CDIM;
    const float* brow1 = bdesc + (size_t)(j0 + 16 + lr) * CDIM;
    const float* brow2 = bdesc + (size_t)(j0 + 32 + lr) * CDIM;
    const float* brow3 = bdesc + (size_t)(j0 + 48 + lr) * CDIM;

    v8f acc0 = {}, acc1 = {}, acc2 = {}, acc3 = {};

    #pragma unroll
    for (int k = 0; k < CDIM; k += 32) {
        int k0 = k + hi * 8;
        v16h af = htl_load_frag_f32(arow, k0);
        acc0 = __builtin_amdgcn_wmma_f32_16x16x32_f16(false, af, false, htl_load_frag_f32(brow0, k0),
                                                      (short)0, acc0, false, false);
        acc1 = __builtin_amdgcn_wmma_f32_16x16x32_f16(false, af, false, htl_load_frag_f32(brow1, k0),
                                                      (short)0, acc1, false, false);
        acc2 = __builtin_amdgcn_wmma_f32_16x16x32_f16(false, af, false, htl_load_frag_f32(brow2, k0),
                                                      (short)0, acc2, false, false);
        acc3 = __builtin_amdgcn_wmma_f32_16x16x32_f16(false, af, false, htl_load_frag_f32(brow3, k0),
                                                      (short)0, acc3, false, false);
    }
    htl_epilogue(acc0, acc1, acc2, acc3, a_sq, b_sq, wkp1, kp2, neg_bits, i0, j0, hi, lr);
}

// ---------------------------------------------------------------------------
// Positive branch: bilinear-sample desc2 at w_kp1/8-0.5, L2-normalize,
// positive_dist[i] = || kp1_desc[i] - pos + 1e-6 ||.  Block = keypoint.
// ---------------------------------------------------------------------------
__global__ __launch_bounds__(256) void htl_positive(
    const float* __restrict__ desc2, const float* __restrict__ wkp1,
    const float* __restrict__ adesc, float* __restrict__ pos_dist) {
    int i = blockIdx.x, c = threadIdx.x;
    float y = wkp1[2 * i] * 0.125f - 0.5f;
    float x = wkp1[2 * i + 1] * 0.125f - 0.5f;
    y = fminf(fmaxf(y, 0.0f), (float)(HC - 1));
    x = fminf(fmaxf(x, 0.0f), (float)(WC - 1));
    float y0f = floorf(y), x0f = floorf(x);
    int y0 = (int)y0f, x0 = (int)x0f;
    int y1 = min(y0 + 1, HC - 1), x1 = min(x0 + 1, WC - 1);
    float wy = y - y0f, wx = x - x0f;
    const float* base = desc2 + (size_t)c * (HC * WC);
    float v00 = base[y0 * WC + x0], v01 = base[y0 * WC + x1];
    float v10 = base[y1 * WC + x0], v11 = base[y1 * WC + x1];
    float v = (1.f - wy) * ((1.f - wx) * v00 + wx * v01)
            +        wy  * ((1.f - wx) * v10 + wx * v11);

    __shared__ float red[256];
    red[c] = v * v;
    __syncthreads();
    for (int s = 128; s > 0; s >>= 1) {
        if (c < s) red[c] += red[c + s];
        __syncthreads();
    }
    float nrm = fmaxf(sqrtf(red[0]), 1e-12f);
    __syncthreads();

    float diff = adesc[(size_t)i * CDIM + c] - v / nrm + 1e-6f;
    red[c] = diff * diff;
    __syncthreads();
    for (int s = 128; s > 0; s >>= 1) {
        if (c < s) red[c] += red[c + s];
        __syncthreads();
    }
    if (c == 0) pos_dist[i] = sqrtf(red[0]);
}

// ---------------------------------------------------------------------------
// Final mean of relu(pos - neg + margin). Single block.
// ---------------------------------------------------------------------------
__global__ __launch_bounds__(1024) void htl_final(const float* __restrict__ pos,
                                                  const unsigned int* __restrict__ negb,
                                                  float* __restrict__ out, int N) {
    __shared__ float red[1024];
    float s = 0.f;
    for (int i = threadIdx.x; i < N; i += 1024) {
        float t = pos[i] - __uint_as_float(negb[i]) + 1.0f;  // MARGIN = 1
        s += (t > 0.f) ? t : 0.f;
    }
    red[threadIdx.x] = s;
    __syncthreads();
    for (int st = 512; st > 0; st >>= 1) {
        if (threadIdx.x < st) red[threadIdx.x] += red[threadIdx.x + st];
        __syncthreads();
    }
    if (threadIdx.x == 0) out[0] = red[0] / (float)N;  // LOSS_LAMBDA = 1
}

// ---------------------------------------------------------------------------
extern "C" void kernel_launch(void* const* d_in, const int* in_sizes, int n_in,
                              void* d_out, int out_size, void* d_ws, size_t ws_size,
                              hipStream_t stream) {
    (void)n_in; (void)out_size;
    const float* kp1      = (const float*)d_in[0]; (void)kp1;  // unused by reference
    const float* w_kp1    = (const float*)d_in[1];
    const float* kp2      = (const float*)d_in[2];
    const float* kp1_desc = (const float*)d_in[3];
    const float* kp2_desc = (const float*)d_in[4];
    const float* desc2    = (const float*)d_in[5];
    const int N = in_sizes[1] / 2;   // 8192
    const int M = in_sizes[2] / 2;   // 8192
    float* out = (float*)d_out;

    const size_t halfBytes  = (size_t)(N + M) * CDIM * sizeof(_Float16);  // f16 staging
    const size_t floatBytes = sizeof(float) * (size_t)(3 * N + M);        // a_sq,b_sq,neg,pos
    char* ws = (char*)d_ws;

    if (ws_size >= halfBytes + floatBytes) {
        // ---- staged path: f16 descriptor copies + async-LDS WMMA GEMM ----
        _Float16* ah = (_Float16*)ws;
        _Float16* bh = (_Float16*)(ws + (size_t)N * CDIM * sizeof(_Float16));
        char* fs = ws + halfBytes;
        float*        a_sq     = (float*)fs;
        float*        b_sq     = (float*)(fs + sizeof(float) * (size_t)N);
        unsigned int* neg_bits = (unsigned int*)(fs + sizeof(float) * (size_t)(N + M));
        float*        pos_dist = (float*)(fs + sizeof(float) * (size_t)(2 * N + M));

        long nA = (long)N * CDIM, nB = (long)M * CDIM;
        htl_f32_to_f16<<<(int)((nA / 8 + 255) / 256), 256, 0, stream>>>(kp1_desc, ah, nA);
        htl_f32_to_f16<<<(int)((nB / 8 + 255) / 256), 256, 0, stream>>>(kp2_desc, bh, nB);
        htl_init<<<(N + 255) / 256, 256, 0, stream>>>(neg_bits, N);
        htl_sqnorm<<<(N + M + 7) / 8, 256, 0, stream>>>(kp1_desc, kp2_desc, a_sq, b_sq, N, M);

        dim3 g(M / 64, N / 64);
        htl_gemm_min_lds<<<g, 128, 0, stream>>>(ah, bh, a_sq, b_sq, w_kp1, kp2, neg_bits);

        htl_positive<<<N, 256, 0, stream>>>(desc2, w_kp1, kp1_desc, pos_dist);
        htl_final<<<1, 1024, 0, stream>>>(pos_dist, neg_bits, out, N);
    } else {
        // ---- fallback: convert-in-loop GEMM, tiny scratch ----
        float*        a_sq     = (float*)ws;
        float*        b_sq     = (float*)(ws + sizeof(float) * (size_t)N);
        unsigned int* neg_bits = (unsigned int*)(ws + sizeof(float) * (size_t)(N + M));
        float*        pos_dist = (float*)(ws + sizeof(float) * (size_t)(2 * N + M));

        htl_init<<<(N + 255) / 256, 256, 0, stream>>>(neg_bits, N);
        htl_sqnorm<<<(N + M + 7) / 8, 256, 0, stream>>>(kp1_desc, kp2_desc, a_sq, b_sq, N, M);

        dim3 g(M / 64, N / 64);
        htl_gemm_min_f<<<g, 128, 0, stream>>>(kp1_desc, kp2_desc, a_sq, b_sq,
                                              w_kp1, kp2, neg_bits);

        htl_positive<<<N, 256, 0, stream>>>(desc2, w_kp1, kp1_desc, pos_dist);
        htl_final<<<1, 1024, 0, stream>>>(pos_dist, neg_bits, out, N);
    }
}
